// MultiHeadedAttention_58514634441299
// MI455X (gfx1250) — matrix operand order repaired and checked by reading, not compiled, not run on hardware
//
#include <hip/hip_runtime.h>

// ---------------------------------------------------------------------------
// MHA forward for MI455X (gfx1250, wave32, WMMA 16x16x32 f16 -> f32 acc)
// B=2, S=2048, D=2048, H=16, DK=128
// GEMMs use async global->LDS double-buffered tiles (ASYNCcnt path).
// ---------------------------------------------------------------------------

typedef _Float16 h8  __attribute__((ext_vector_type(8)));
typedef _Float16 h16 __attribute__((ext_vector_type(16)));
typedef float    f8  __attribute__((ext_vector_type(8)));

__device__ __forceinline__ h16 mk16(h8 lo, h8 hi) {
  return __builtin_shufflevector(lo, hi, 0,1,2,3,4,5,6,7,8,9,10,11,12,13,14,15);
}

// A-fragment (16xK=32, f16): lane (r=lane&15) holds K runs [8h..8h+7] and [16+8h..+7]
__device__ __forceinline__ h16 fragA(const _Float16* prow, int halfid) {
  h8 lo = *(const h8*)(prow + (halfid << 3));
  h8 hi = *(const h8*)(prow + 16 + (halfid << 3));
  return mk16(lo, hi);
}
// B-fragment (K=32 x 16, f16): lane (col=lane&15) holds K runs [8h..8h+7] and [16+8h..+7]
__device__ __forceinline__ h16 fragB(const _Float16* prow, int halfid) {
  h8 lo = *(const h8*)(prow + (halfid << 3));
  h8 hi = *(const h8*)(prow + 16 + (halfid << 3));
  return mk16(lo, hi);
}

__device__ __forceinline__ f8 wmma(h16 a, h16 b, f8 c) {
  return __builtin_amdgcn_wmma_f32_16x16x32_f16(false, a, false, b, (short)0, c,
                                                false, false);
}

// ---------------------------------------------------------------------------
// f32 -> f16 elementwise convert
// ---------------------------------------------------------------------------
__global__ __launch_bounds__(256) void cvt_f32_f16_k(const float* __restrict__ s,
                                                     _Float16* __restrict__ d, int n) {
  int i = blockIdx.x * 256 + threadIdx.x;
  if (i < n) d[i] = (_Float16)s[i];
}

// ---------------------------------------------------------------------------
// GEMM: out = A[M,K] * W[N,K]^T + bias[N]    (M=4096, N=2048, K=2048)
// block = 128 thr (4 waves, 2x2), block tile 64x128, wave tile 32x64.
// A (64x32) and W (128x32) k-slices staged in LDS via
// GLOBAL_LOAD_ASYNC_TO_LDS_B128, double-buffered, ASYNCcnt-synchronized.
// Rows padded to 40 f16 (80B = 5x16B chunks -> all frag loads 16B aligned).
// MODE 0: f16 out [B,H,S,DK] (Q,K)   MODE 1: f16 out [B,H,DK,S] (V, transposed)
// MODE 2: f32 out [M,N] (output projection -> d_out)
// ---------------------------------------------------------------------------
template <int MODE>
__global__ __launch_bounds__(128) void gemm_nt_k(const _Float16* __restrict__ A,
                                                 const _Float16* __restrict__ W,
                                                 const float* __restrict__ bias,
                                                 void* __restrict__ dst) {
  __shared__ _Float16 At[2][64 * 40];
  __shared__ _Float16 Wt[2][128 * 40];

  const int tid = threadIdx.x;
  const int lane = tid & 31, wv = tid >> 5;
  const int halfid = lane >> 4, r = lane & 15;
  const int mB = blockIdx.x * 64;    // block row base (tokens)
  const int nB = blockIdx.y * 128;   // block col base (features)
  const int mw = (wv >> 1) * 32;     // wave row offset in tile
  const int nw = (wv & 1) * 64;      // wave col offset in tile

  const unsigned long long Ag = (unsigned long long)(uintptr_t)A;
  const unsigned long long Wg = (unsigned long long)(uintptr_t)W;

  // async fill of one double-buffer slot with the k-slice starting at kb
  auto fill = [&](int buf, int kb) {
    // W tile: 128 rows x 32 f16 = 4 chunks of 16B per row, 4 copies/thread
#pragma unroll
    for (int j = 0; j < 4; ++j) {
      const int id = j * 128 + tid, row = id >> 2, ch = id & 3;
      const unsigned voff =
          (unsigned)(((nB + row) * 2048 + kb) * 2 + ch * 16);   // bytes
      const unsigned ldst =
          (unsigned)(uintptr_t)&Wt[buf][row * 40] + (unsigned)(ch * 16);
      asm volatile("global_load_async_to_lds_b128 %0, %1, %2"
                   :: "v"(ldst), "v"(voff), "s"(Wg) : "memory");
    }
    // A tile: 64 rows x 32 f16, 2 copies/thread
#pragma unroll
    for (int j = 0; j < 2; ++j) {
      const int id = j * 128 + tid, row = id >> 2, ch = id & 3;
      const unsigned voff =
          (unsigned)(((mB + row) * 2048 + kb) * 2 + ch * 16);
      const unsigned ldst =
          (unsigned)(uintptr_t)&At[buf][row * 40] + (unsigned)(ch * 16);
      asm volatile("global_load_async_to_lds_b128 %0, %1, %2"
                   :: "v"(ldst), "v"(voff), "s"(Ag) : "memory");
    }
  };

  f8 c[2][4] = {};

  fill(0, 0);
  asm volatile("s_wait_asynccnt 0" ::: "memory");
  __syncthreads();

  for (int kb = 0; kb < 2048; kb += 32) {
    const int cur = (kb >> 5) & 1;
    // prefetch next k-slice into the other buffer (wraps harmlessly at the end)
    fill(cur ^ 1, (kb + 32) & 2047);

    h16 a0 = fragA(&At[cur][(mw + r) * 40], halfid);
    h16 a1 = fragA(&At[cur][(mw + 16 + r) * 40], halfid);
#pragma unroll
    for (int nt = 0; nt < 4; ++nt) {
      h16 b = fragB(&Wt[cur][(nw + nt * 16 + r) * 40], halfid);
      c[0][nt] = wmma(a0, b, c[0][nt]);
      c[1][nt] = wmma(a1, b, c[1][nt]);
    }
    // prefetch done + all DS reads of this buffer retired before reuse
    asm volatile("s_wait_asynccnt 0\n\ts_wait_dscnt 0" ::: "memory");
    __syncthreads();
  }

#pragma unroll
  for (int mt = 0; mt < 2; ++mt)
#pragma unroll
    for (int nt = 0; nt < 4; ++nt) {
      const int n = nB + nw + nt * 16 + r;
      const float bn = bias[n];
#pragma unroll
      for (int g = 0; g < 8; ++g) {
        const int m = mB + mw + mt * 16 + g + (halfid << 3);
        const float v = c[mt][nt][g] + bn;
        if (MODE == 2) {
          ((float*)dst)[(size_t)m * 2048 + n] = v;
        } else {
          const int bb = m >> 11, s = m & 2047, h = n >> 7, d = n & 127;
          const size_t idx =
              (MODE == 0) ? ((((size_t)bb * 16 + h) * 2048 + s) * 128 + d)
                          : ((((size_t)bb * 16 + h) * 128 + d) * 2048 + s);
          ((_Float16*)dst)[idx] = (_Float16)v;
        }
      }
    }
}

// ---------------------------------------------------------------------------
// RoPE in place on [B,H,S,DK] f16; pairs (d, d+64); Q also gets 1/sqrt(DK)
// ---------------------------------------------------------------------------
__global__ __launch_bounds__(256) void rope_k(_Float16* __restrict__ buf, float scale) {
  int i = blockIdx.x * 256 + threadIdx.x;      // 2*16*2048*64 threads
  if (i >= 2 * 16 * 2048 * 64) return;
  const int d = i & 63, s = (i >> 6) & 2047, bh = i >> 17;
  _Float16* p = buf + ((size_t)bh * 2048 + s) * 128 + d;
  const float x0 = (float)p[0], x1 = (float)p[64];
  // inv_freq = 10000^(-d/64) = 2^(-d*log2(10000)/64)
  const float inv = exp2f(-(float)d * (13.287712379549449f / 64.0f));
  float sn, cs;
  sincosf((float)s * inv, &sn, &cs);
  p[0]  = (_Float16)((x0 * cs - x1 * sn) * scale);
  p[64] = (_Float16)((x1 * cs + x0 * sn) * scale);
}

// ---------------------------------------------------------------------------
// Flash attention: Q[B,H,S,DK] f16 (pre-scaled), K[B,H,S,DK] f16, Vt[B,H,DK,S] f16
// -> O[B,S,D] f16. grid (B*H, S/64), block 128 (4 waves, 16 q-rows each).
// Per-wave causal trip counts -> no block barriers; LDS bounce is wave-private.
// ---------------------------------------------------------------------------
__global__ __launch_bounds__(128) void flash_k(const _Float16* __restrict__ Q,
                                               const _Float16* __restrict__ Kt,
                                               const _Float16* __restrict__ Vt,
                                               _Float16* __restrict__ O) {
  __shared__ _Float16 plds[4][16][72];   // per-wave 16x64 P tile, padded rows
  const int lane = threadIdx.x & 31, wv = threadIdx.x >> 5;
  const int halfid = lane >> 4, r = lane & 15;
  const int bh = blockIdx.x, b = bh >> 4, h = bh & 15;
  const int q0 = blockIdx.y * 64 + wv * 16;

  const _Float16* qbase = Q + ((size_t)bh * 2048 + q0 + r) * 128;
  h16 aq[4];
#pragma unroll
  for (int kc = 0; kc < 4; ++kc) aq[kc] = fragA(qbase + kc * 32, halfid);

  f8 o[8] = {};
  float mi[8], li[8];
#pragma unroll
  for (int g = 0; g < 8; ++g) { mi[g] = -1e30f; li[g] = 0.0f; }

  for (int kv0 = 0; kv0 < q0 + 16; kv0 += 64) {   // wave-uniform causal bound
    f8 sc[4];
#pragma unroll
    for (int nt = 0; nt < 4; ++nt) {
      f8 acc = {};
      const _Float16* kbase = Kt + ((size_t)bh * 2048 + kv0 + nt * 16 + r) * 128;
#pragma unroll
      for (int kc = 0; kc < 4; ++kc)
        acc = wmma(aq[kc], fragB(kbase + kc * 32, halfid), acc);
      sc[nt] = acc;
    }
    if (kv0 + 63 > q0) {   // causal mask, uniform branch
#pragma unroll
      for (int nt = 0; nt < 4; ++nt)
#pragma unroll
        for (int g = 0; g < 8; ++g) {
          const int qi = q0 + g + (halfid << 3);
          const int kj = kv0 + nt * 16 + r;
          if (kj > qi) sc[nt][g] = -1e30f;
        }
    }
    // online softmax: C-tile rows live in one 16-lane half; xor masks 1/2/4/8
    // never cross the half boundary.
    float alpha[8];
#pragma unroll
    for (int g = 0; g < 8; ++g) {
      float t = fmaxf(fmaxf(sc[0][g], sc[1][g]), fmaxf(sc[2][g], sc[3][g]));
#pragma unroll
      for (int mk = 1; mk < 16; mk <<= 1) t = fmaxf(t, __shfl_xor(t, mk, 32));
      const float mnew = fmaxf(mi[g], t);
      alpha[g] = __expf(mi[g] - mnew);
      float rs = 0.0f;
#pragma unroll
      for (int nt = 0; nt < 4; ++nt) {
        const float e = __expf(sc[nt][g] - mnew);
        sc[nt][g] = e;
        rs += e;
      }
#pragma unroll
      for (int mk = 1; mk < 16; mk <<= 1) rs += __shfl_xor(rs, mk, 32);
      li[g] = li[g] * alpha[g] + rs;
      mi[g] = mnew;
    }
#pragma unroll
    for (int ot = 0; ot < 8; ++ot)
#pragma unroll
      for (int g = 0; g < 8; ++g) o[ot][g] *= alpha[g];

    // P (f32, C-layout) -> f16 in LDS -> reload as A-fragments
#pragma unroll
    for (int nt = 0; nt < 4; ++nt)
#pragma unroll
      for (int g = 0; g < 8; ++g)
        plds[wv][g + (halfid << 3)][nt * 16 + r] = (_Float16)sc[nt][g];
    asm volatile("s_wait_dscnt 0" ::: "memory");  // cross-lane LDS dependency
    h16 pa0 = fragA(&plds[wv][r][0], halfid);
    h16 pa1 = fragA(&plds[wv][r][32], halfid);

    // O += P * V   (V transposed [B,H,DK,S]: B-fragments are K-contiguous)
#pragma unroll
    for (int ot = 0; ot < 8; ++ot) {
      const _Float16* vbase = Vt + ((size_t)bh * 128 + ot * 16 + r) * 2048 + kv0;
      o[ot] = wmma(pa0, fragB(vbase, halfid), o[ot]);
      o[ot] = wmma(pa1, fragB(vbase + 32, halfid), o[ot]);
    }
  }

#pragma unroll
  for (int ot = 0; ot < 8; ++ot)
#pragma unroll
    for (int g = 0; g < 8; ++g) {
      const int qi = q0 + g + (halfid << 3);
      const int d = ot * 16 + r;
      const float v = o[ot][g] / li[g];
      O[((size_t)b * 2048 + qi) * 2048 + (size_t)h * 128 + d] = (_Float16)v;
    }
}

// ---------------------------------------------------------------------------
// Host launcher
// ---------------------------------------------------------------------------
extern "C" void kernel_launch(void* const* d_in, const int* in_sizes, int n_in,
                              void* d_out, int out_size, void* d_ws, size_t ws_size,
                              hipStream_t stream) {
  (void)in_sizes; (void)n_in; (void)out_size;
  const float* q  = (const float*)d_in[0];
  const float* k  = (const float*)d_in[1];
  const float* v  = (const float*)d_in[2];
  // d_in[3] = mask (causal, known analytically)
  const float* wq = (const float*)d_in[4];  const float* bq = (const float*)d_in[5];
  const float* wk = (const float*)d_in[6];  const float* bk = (const float*)d_in[7];
  const float* wv = (const float*)d_in[8];  const float* bv = (const float*)d_in[9];
  const float* wo = (const float*)d_in[10]; const float* bo = (const float*)d_in[11];
  float* out = (float*)d_out;

  const size_t SZX = (size_t)4096 * 2048;   // B*S x D
  const size_t SZW = (size_t)2048 * 2048;   // D x D
  if (ws_size < (7 * SZX + 4 * SZW) * sizeof(_Float16)) return;

  _Float16* xq   = (_Float16*)d_ws;
  _Float16* xk   = xq + SZX;
  _Float16* xv   = xk + SZX;
  _Float16* wq16 = xv + SZX;
  _Float16* wk16 = wq16 + SZW;
  _Float16* wv16 = wk16 + SZW;
  _Float16* wo16 = wv16 + SZW;
  _Float16* Qh   = wo16 + SZW;   // [B,H,S,DK]
  _Float16* Kh   = Qh + SZX;     // [B,H,S,DK]
  _Float16* VhT  = Kh + SZX;     // [B,H,DK,S]
  _Float16* attn = VhT + SZX;    // [B,S,D]

  // 1) f32 -> f16 conversions
  cvt_f32_f16_k<<<SZX / 256, 256, 0, stream>>>(q, xq, (int)SZX);
  cvt_f32_f16_k<<<SZX / 256, 256, 0, stream>>>(k, xk, (int)SZX);
  cvt_f32_f16_k<<<SZX / 256, 256, 0, stream>>>(v, xv, (int)SZX);
  cvt_f32_f16_k<<<SZW / 256, 256, 0, stream>>>(wq, wq16, (int)SZW);
  cvt_f32_f16_k<<<SZW / 256, 256, 0, stream>>>(wk, wk16, (int)SZW);
  cvt_f32_f16_k<<<SZW / 256, 256, 0, stream>>>(wv, wv16, (int)SZW);
  cvt_f32_f16_k<<<SZW / 256, 256, 0, stream>>>(wo, wo16, (int)SZW);

  // 2) QKV projections (WMMA + async LDS tiles)
  dim3 ggrid(64, 16), gblk(128);
  gemm_nt_k<0><<<ggrid, gblk, 0, stream>>>(xq, wq16, bq, Qh);
  gemm_nt_k<0><<<ggrid, gblk, 0, stream>>>(xk, wk16, bk, Kh);
  gemm_nt_k<1><<<ggrid, gblk, 0, stream>>>(xv, wv16, bv, VhT);

  // 3) RoPE (fold 1/sqrt(DK) into Q)
  const int nrope = 2 * 16 * 2048 * 64;
  rope_k<<<nrope / 256, 256, 0, stream>>>(Qh, 0.08838834764831845f);
  rope_k<<<nrope / 256, 256, 0, stream>>>(Kh, 1.0f);

  // 4) causal flash attention (WMMA)
  flash_k<<<dim3(32, 32), 128, 0, stream>>>(Qh, Kh, VhT, attn);

  // 5) output projection -> f32 d_out
  gemm_nt_k<2><<<ggrid, gblk, 0, stream>>>(attn, wo16, bo, out);
}
